// SparseTransform_2327872274826
// MI455X (gfx1250) — compile-verified, matrix operand order checked by reading, not run on hardware
//
#include <hip/hip_runtime.h>
#include <hip/hip_bf16.h>
#include <stdint.h>
#include <math.h>

// ---------------- problem constants ----------------
#define B_   16
#define N_   250000
#define FLIP_RATIO 0.25f
#define EPS_ 0.001f

// ---------------- workspace layout (float/uint indices into d_ws) ----------
#define P_PARAMS 0     // 16 * 6 floats : flip, cos, sin, shift0..2
#define P_R      96    // 16 * 6 floats : r1[3], r2[3]
#define P_SUMS   192   // 16 * 8 floats : sum(x,y,z,w), sumsq(x,y,z,w)
#define P_MM     320   // 16 * 6 uints  : enc-min[3], enc-max[3]
#define P_MS     416   // 16 * 8 floats : mean[4], inv_std[4]
#define P_OFF    544   // 16 * 3 floats : offset[3]

// ---------------- d_out layout (flat float, return order) ------------------
#define OUT_FEATS 16000000LL   // locs (B,N,4) first
#define OUT_ORG   32000000LL
#define OUT_LBL   60000000LL
#define OUT_NM    64000000LL
#define OUT_OFFS  68000000LL

typedef __attribute__((ext_vector_type(16))) _Float16 v16h;
typedef __attribute__((ext_vector_type(8)))  float    v8f;

// ---------------- threefry2x32 (jax-compatible schedule) --------------------
__device__ __forceinline__ void tf2x32(uint32_t k0, uint32_t k1, uint32_t c0, uint32_t c1,
                                       uint32_t& o0, uint32_t& o1) {
  const uint32_t R0[4] = {13u, 15u, 26u, 6u};
  const uint32_t R1[4] = {17u, 29u, 16u, 24u};
  uint32_t ks[3] = {k0, k1, 0x1BD11BDAu ^ k0 ^ k1};
  uint32_t x0 = c0 + ks[0], x1 = c1 + ks[1];
  for (uint32_t g = 0u; g < 5u; ++g) {
    const uint32_t* R = (g & 1u) ? R1 : R0;
    for (int r = 0; r < 4; ++r) {
      x0 += x1;
      x1 = (x1 << R[r]) | (x1 >> (32u - R[r]));
      x1 ^= x0;
    }
    x0 += ks[(g + 1u) % 3u];
    x1 += ks[(g + 2u) % 3u] + (g + 1u);
  }
  o0 = x0; o1 = x1;
}

__device__ __forceinline__ float u01(uint32_t bits) {
  return (float)(bits >> 8) * (1.0f / 16777216.0f);
}

// monotone uint encoding of float for atomicMin/Max
__device__ __forceinline__ unsigned fenc(float f) {
  unsigned u = __float_as_uint(f);
  return (u & 0x80000000u) ? ~u : (u | 0x80000000u);
}
__device__ __forceinline__ float fdec(unsigned e) {
  unsigned u = (e & 0x80000000u) ? (e & 0x7FFFFFFFu) : ~e;
  return __uint_as_float(u);
}

// ---------------- kernel 1: per-sample RNG + accumulator reset --------------
__global__ void st_setup(float* __restrict__ wsf, unsigned* __restrict__ wsu) {
  int t = threadIdx.x;
  for (int i = t; i < B_ * 8; i += 32) wsf[P_SUMS + i] = 0.0f;
  for (int i = t; i < B_ * 6; i += 32) wsu[P_MM + i] = ((i % 6) < 3) ? 0xFFFFFFFFu : 0u;
  if (t < B_) {
    uint32_t keys[5][2];
    for (int j = 0; j < 5; ++j) {
      uint32_t idx = (uint32_t)(t * 5 + j);
      tf2x32(0u, 42u, 2u * idx, 2u * idx + 1u, keys[j][0], keys[j][1]);
    }
    uint32_t a, b, a2, b2;
    // flip
    tf2x32(keys[0][0], keys[0][1], 0u, 0u, a, b);
    float flip = (u01(a) < FLIP_RATIO) ? 1.0f : 0.0f;
    // rotation angle
    tf2x32(keys[1][0], keys[1][1], 0u, 0u, a, b);
    float ang = (u01(a) - 0.5f) * (30.0f / 180.0f) * 3.14159265358979f;
    float cv = cosf(ang), sv = sinf(ang);
    // shift: 3 normals via Box-Muller
    tf2x32(keys[2][0], keys[2][1], 0u, 0u, a, b);
    tf2x32(keys[2][0], keys[2][1], 1u, 0u, a2, b2);
    float u1 = fmaxf(u01(a), 1e-7f), u2 = u01(b);
    float u3 = fmaxf(u01(a2), 1e-7f), u4 = u01(b2);
    float m1 = sqrtf(-2.0f * logf(u1));
    float m2 = sqrtf(-2.0f * logf(u3));
    const float TWO_PI = 6.28318530717958647f;
    float n0 = m1 * cosf(TWO_PI * u2);
    float n1 = m1 * sinf(TWO_PI * u2);
    float n2 = m2 * cosf(TWO_PI * u4);
    wsf[P_PARAMS + t * 6 + 0] = flip;
    wsf[P_PARAMS + t * 6 + 1] = cv;
    wsf[P_PARAMS + t * 6 + 2] = sv;
    wsf[P_PARAMS + t * 6 + 3] = n0 * 0.10f;
    wsf[P_PARAMS + t * 6 + 4] = n1 * 0.10f;
    wsf[P_PARAMS + t * 6 + 5] = n2 * 0.05f;
    // r1, r2 uniforms
    tf2x32(keys[3][0], keys[3][1], 0u, 0u, a, b);
    tf2x32(keys[3][0], keys[3][1], 1u, 0u, a2, b2);
    wsf[P_R + t * 6 + 0] = u01(a);
    wsf[P_R + t * 6 + 1] = u01(b);
    wsf[P_R + t * 6 + 2] = u01(a2);
    tf2x32(keys[4][0], keys[4][1], 0u, 0u, a, b);
    tf2x32(keys[4][0], keys[4][1], 1u, 0u, a2, b2);
    wsf[P_R + t * 6 + 3] = u01(a);
    wsf[P_R + t * 6 + 4] = u01(b);
    wsf[P_R + t * 6 + 5] = u01(a2);
  }
}

// ---------------- kernel 2: streaming reduction (sum/sumsq/min/max) ---------
// Cross-lane column sums done with one v_wmma_f32_16x16x32_f16 per wave:
// A = ones(16x32), B(32x16) column q = the 32 lane-partials of quantity q
// (transposed through LDS, cols 8..15 zero-filled so the per-lane fragment
// read is one contiguous, unconditional 32B LDS load -> ds_load_b128 pair).
__global__ __launch_bounds__(256) void st_reduce(const float4* __restrict__ pts,
                                                 const int* __restrict__ scale_p,
                                                 float* __restrict__ wsf,
                                                 unsigned* __restrict__ wsu) {
  const int b    = blockIdx.y;
  const int w    = threadIdx.x >> 5;
  const int lane = threadIdx.x & 31;

  const float flip = wsf[P_PARAMS + b * 6 + 0];
  const float cv   = wsf[P_PARAMS + b * 6 + 1];
  const float sv   = wsf[P_PARAMS + b * 6 + 2];
  const float sh0  = wsf[P_PARAMS + b * 6 + 3];
  const float sh1  = wsf[P_PARAMS + b * 6 + 4];
  const float sh2  = wsf[P_PARAMS + b * 6 + 5];
  const float ysgn = 1.0f - 2.0f * flip;
  const float sc   = (float)scale_p[0];

  float sum0 = 0.f, sum1 = 0.f, sum2 = 0.f, sum3 = 0.f;
  float sq0 = 0.f, sq1 = 0.f, sq2 = 0.f, sq3 = 0.f;
  float mn0 = INFINITY, mn1 = INFINITY, mn2 = INFINITY;
  float mx0 = -INFINITY, mx1 = -INFINITY, mx2 = -INFINITY;

  const int stride = blockDim.x * gridDim.x;
  for (int i = blockIdx.x * blockDim.x + threadIdx.x; i < N_; i += stride) {
    const float4* cur = &pts[(size_t)b * N_ + i];
    __builtin_prefetch(cur + stride, 0, 3);   // global_prefetch_b8, speculative
    float4 p = *cur;
    float x = p.x;
    float y = p.y * ysgn;
    float xr = x * cv - y * sv + sh0;
    float yr = x * sv + y * cv + sh1;
    float z  = p.z + sh2;
    float wv = p.w;
    sum0 += xr; sum1 += yr; sum2 += z; sum3 += wv;
    sq0 += xr * xr; sq1 += yr * yr; sq2 += z * z; sq3 += wv * wv;
    float a0 = sc * xr, a1 = sc * yr, a2 = sc * z;
    mn0 = fminf(mn0, a0); mn1 = fminf(mn1, a1); mn2 = fminf(mn2, a2);
    mx0 = fmaxf(mx0, a0); mx1 = fmaxf(mx1, a1); mx2 = fmaxf(mx2, a2);
  }

  // ---- min/max: wave32 shuffle tree, then encoded global atomics ----
  for (int off = 16; off > 0; off >>= 1) {
    mn0 = fminf(mn0, __shfl_xor(mn0, off, 32));
    mn1 = fminf(mn1, __shfl_xor(mn1, off, 32));
    mn2 = fminf(mn2, __shfl_xor(mn2, off, 32));
    mx0 = fmaxf(mx0, __shfl_xor(mx0, off, 32));
    mx1 = fmaxf(mx1, __shfl_xor(mx1, off, 32));
    mx2 = fmaxf(mx2, __shfl_xor(mx2, off, 32));
  }
  if (lane == 0) {
    atomicMin(&wsu[P_MM + b * 6 + 0], fenc(mn0));
    atomicMin(&wsu[P_MM + b * 6 + 1], fenc(mn1));
    atomicMin(&wsu[P_MM + b * 6 + 2], fenc(mn2));
    atomicMax(&wsu[P_MM + b * 6 + 3], fenc(mx0));
    atomicMax(&wsu[P_MM + b * 6 + 4], fenc(mx1));
    atomicMax(&wsu[P_MM + b * 6 + 5], fenc(mx2));
  }

  // ---- sums: WMMA column-sum reduction across the wave ----
  // xp[w][col][k] : col = B-matrix column (quantity), k = K index (lane id)
  __shared__ _Float16 xp[8][16][32];
  float part[8] = {sum0, sum1, sum2, sum3, sq0, sq1, sq2, sq3};
  #pragma unroll
  for (int q = 0; q < 8; ++q) {
    xp[w][q][lane]     = (_Float16)part[q];
    xp[w][q + 8][lane] = (_Float16)0.0f;     // zero-fill unused columns
  }
  __syncthreads();

  // lane L holds column N = L&15, K = (L>=16 ? 16..31 : 0..15):
  // a single contiguous 32-byte, 32B-aligned LDS read -> no divergence.
  const v16h bfrag = *(const v16h*)&xp[w][lane & 15][(lane >> 4) << 4];
  v16h aones;
  #pragma unroll
  for (int i = 0; i < 16; ++i) aones[i] = (_Float16)1.0f;
  v8f cacc = {};
  cacc = __builtin_amdgcn_wmma_f32_16x16x32_f16(
      /*neg_a=*/false, aones, /*neg_b=*/false, bfrag,
      /*c_mod=*/(short)0, cacc, /*reuse_a=*/false, /*reuse_b=*/false);
  // D VGPR0, lanes 0..7 hold (M=0, N=lane) = total of quantity `lane`
  if (lane < 8) atomicAdd(&wsf[P_SUMS + b * 8 + lane], cacc[0]);
}

// ---------------- kernel 3: per-sample finalize (mean/std/offset) -----------
__global__ void st_finalize(float* __restrict__ wsf, const unsigned* __restrict__ wsu,
                            const float* __restrict__ full_scale, float* __restrict__ out) {
  int b = threadIdx.x;
  if (b >= B_) return;
  const float Nf = (float)N_;
  #pragma unroll
  for (int q = 0; q < 4; ++q) {
    float s  = wsf[P_SUMS + b * 8 + q];
    float ss = wsf[P_SUMS + b * 8 + 4 + q];
    float mean = s / Nf;
    float var  = (ss - s * s / Nf) / (Nf - 1.0f);
    wsf[P_MS + b * 8 + q]     = mean;
    wsf[P_MS + b * 8 + 4 + q] = rsqrtf(fmaxf(var, 1e-20f));
  }
  #pragma unroll
  for (int k = 0; k < 3; ++k) {
    float m  = fdec(wsu[P_MM + b * 6 + k]);
    float M  = fdec(wsu[P_MM + b * 6 + 3 + k]);
    float qr = M - m;
    float fs = full_scale[k];
    float r1 = wsf[P_R + b * 6 + k];
    float r2 = wsf[P_R + b * 6 + 3 + k];
    float off = -m + fmaxf(fs - qr - EPS_, 0.0f) * r1 + fminf(fs - qr + EPS_, 0.0f) * r2;
    wsf[P_OFF + b * 3 + k] = off;
    out[OUT_OFFS + b * 3 + k] = off;
  }
}

// ---------------- kernel 4: streaming emission of all outputs ---------------
__global__ __launch_bounds__(256) void st_emit(const float4* __restrict__ pts,
                                               const int* __restrict__ tgt,
                                               const int* __restrict__ msk,
                                               const float* __restrict__ full_scale,
                                               const int* __restrict__ scale_p,
                                               const float* __restrict__ wsf,
                                               float* __restrict__ out) {
  const int b = blockIdx.y;
  const float flip = wsf[P_PARAMS + b * 6 + 0];
  const float cv   = wsf[P_PARAMS + b * 6 + 1];
  const float sv   = wsf[P_PARAMS + b * 6 + 2];
  const float sh0  = wsf[P_PARAMS + b * 6 + 3];
  const float sh1  = wsf[P_PARAMS + b * 6 + 4];
  const float sh2  = wsf[P_PARAMS + b * 6 + 5];
  const float ysgn = 1.0f - 2.0f * flip;
  const float sc   = (float)scale_p[0];
  const float me0 = wsf[P_MS + b * 8 + 0], me1 = wsf[P_MS + b * 8 + 1];
  const float me2 = wsf[P_MS + b * 8 + 2], me3 = wsf[P_MS + b * 8 + 3];
  const float iv0 = wsf[P_MS + b * 8 + 4], iv1 = wsf[P_MS + b * 8 + 5];
  const float iv2 = wsf[P_MS + b * 8 + 6], iv3 = wsf[P_MS + b * 8 + 7];
  const float of0 = wsf[P_OFF + b * 3 + 0], of1 = wsf[P_OFF + b * 3 + 1], of2 = wsf[P_OFF + b * 3 + 2];
  const float fs0 = full_scale[0], fs1 = full_scale[1], fs2 = full_scale[2];

  const int stride = blockDim.x * gridDim.x;
  for (int i = blockIdx.x * blockDim.x + threadIdx.x; i < N_; i += stride) {
    const long long idx = (long long)b * N_ + i;
    const float4* cur = &pts[idx];
    __builtin_prefetch(cur + stride, 0, 3);   // global_prefetch_b8, speculative
    float4 p = *cur;
    float x = p.x;
    float y = p.y * ysgn;
    float xr = x * cv - y * sv + sh0;
    float yr = x * sv + y * cv + sh1;
    float z  = p.z + sh2;
    float wv = p.w;
    float f0 = (xr - me0) * iv0;
    float f1 = (yr - me1) * iv1;
    float f2 = (z  - me2) * iv2;
    float f3 = (wv - me3) * iv3;
    float a0 = sc * xr + of0, a1 = sc * yr + of1, a2 = sc * z + of2;
    int m = msk[idx];
    bool valid = (fminf(fminf(a0, a1), a2) >= 0.0f) &&
                 (a0 < fs0) && (a1 < fs1) && (a2 < fs2) && (m > 0);
    // locs (int-truncated, stored as float), col 3 = batch index
    float4 locv = valid ? make_float4((float)(int)a0, (float)(int)a1, (float)(int)a2, (float)b)
                        : make_float4(0.0f, 0.0f, 0.0f, (float)b);
    ((float4*)out)[idx] = locv;
    // feats (masked)
    float4 fo = valid ? make_float4(f0, f1, f2, f3) : make_float4(0.f, 0.f, 0.f, 0.f);
    *(float4*)(out + OUT_FEATS + idx * 4) = fo;
    // org = [xyz, feat] (unmasked)
    long long ob = OUT_ORG + idx * 7;
    out[ob + 0] = xr; out[ob + 1] = yr; out[ob + 2] = z;
    out[ob + 3] = f0; out[ob + 4] = f1; out[ob + 5] = f2; out[ob + 6] = f3;
    // labels, nmasks
    out[OUT_LBL + idx] = valid ? (float)tgt[idx] : 0.0f;
    out[OUT_NM  + idx] = (float)(m * (valid ? 1 : 0));
  }
}

extern "C" void kernel_launch(void* const* d_in, const int* in_sizes, int n_in,
                              void* d_out, int out_size, void* d_ws, size_t ws_size,
                              hipStream_t stream) {
  (void)in_sizes; (void)n_in; (void)out_size; (void)ws_size;
  const float4* pts       = (const float4*)d_in[0];
  const int*    targets   = (const int*)d_in[1];
  const int*    masks     = (const int*)d_in[2];
  const float*  fullscale = (const float*)d_in[3];
  const int*    scale_p   = (const int*)d_in[4];
  float*    out = (float*)d_out;
  float*    wsf = (float*)d_ws;
  unsigned* wsu = (unsigned*)d_ws;

  st_setup<<<1, 32, 0, stream>>>(wsf, wsu);
  dim3 grid(128, B_);
  st_reduce<<<grid, 256, 0, stream>>>(pts, scale_p, wsf, wsu);
  st_finalize<<<1, B_, 0, stream>>>(wsf, wsu, fullscale, out);
  st_emit<<<grid, 256, 0, stream>>>(pts, targets, masks, fullscale, scale_p, wsf, out);
}
